// Update_78726750535839
// MI455X (gfx1250) — matrix operand order repaired and checked by reading, not compiled, not run on hardware
//
#include <hip/hip_runtime.h>
#include <hip/hip_bf16.h>

// ---------------------------------------------------------------------------
// CDNA5 (gfx1250, wave32) implementation of the DPVO-style update operator.
// All dense layers run through v_wmma_f32_16x16x32_f16 with fused epilogues.
// ---------------------------------------------------------------------------

typedef __attribute__((ext_vector_type(16))) _Float16 v16h;
typedef __attribute__((ext_vector_type(8)))  _Float16 v8h;
typedef __attribute__((ext_vector_type(2)))  _Float16 v2h;
typedef __attribute__((ext_vector_type(8)))  float    v8f;

#define DIM   384
#define NEDGE 4096

union V16U { v16h v; v8h h[2]; };
union F8U  { v8f  v; float f[8]; };

// Block tile: 128(M) x 128(N), K-step 32. 8 waves; wave w owns M rows
// [w*16, w*16+16) across all 128 N columns -> 8 WMMAs per K-step, A reused 8x.
// M multiple of 128 and N multiple of 128 (true for every launch site).
// ACT: 0=none 1=relu 2=sigmoid 3=exp(clip(x,-50,50))
// COMB: 0=none 1=+R1 2=+R1+R2 3=R1+R2*out (gated residual)
template <int GATHER, int KTAIL, int ACT, int COMB>
__global__ __launch_bounds__(256)
void gemm_f16_wmma(const float* __restrict__ X, const int* __restrict__ rowmap,
                   const float* __restrict__ W, const float* __restrict__ bias,
                   const float* __restrict__ R1, const float* __restrict__ R2,
                   float* __restrict__ Y,
                   int M, int K, int N) {
  __shared__ _Float16 lsA[128 * 32];   // 8 KB, row-major [m][k]
  __shared__ _Float16 lsB[128 * 32];   // 8 KB, transposed [n][k]

  const int tid  = threadIdx.x;
  const int lane = tid & 31;
  const int wv   = tid >> 5;       // wave 0..7 -> M sub-tile
  const int hi   = lane >> 4;      // half-wave
  const int l15  = lane & 15;
  const int m0   = blockIdx.y * 128;
  const int n0   = blockIdx.x * 128;

  // A staging geometry: thread handles 8 float2 (row = q/16, col pair q%16)
  const int ar = (tid * 8) >> 4;            // base handled differently below
  (void)ar;

  v8f acc[8];
#pragma unroll
  for (int nt = 0; nt < 8; ++nt) acc[nt] = (v8f){0.f,0.f,0.f,0.f,0.f,0.f,0.f,0.f};

  for (int k0 = 0; k0 < K; k0 += 32) {
    // ---- prefetch next K slice (non-gather path only) -------------------
    if (!GATHER && (k0 + 32) < K) {
      int pr = m0 + (tid >> 1);
      __builtin_prefetch(X + (size_t)pr * K + (k0 + 32), 0, 1);
      int pk = k0 + 32 + (tid & 31);
      if (pk >= K) pk = K - 1;
      __builtin_prefetch(W + (size_t)pk * N + n0 + ((tid >> 5) << 5), 0, 1);
    }

    // ---- stage A tile (128 x 32): load phase then store phase -----------
    float2 av[8];
#pragma unroll
    for (int e = 0; e < 8; ++e) {
      int q  = e * 256 + tid;                   // float2 index 0..2047
      int r  = q >> 4;                          // 16 float2 per row
      int c  = (q & 15) * 2;
      int sr = GATHER ? rowmap[m0 + r] : (m0 + r);
      const float* xr = X + (size_t)sr * K;
      if (KTAIL) {
        av[e].x = (k0 + c     < K) ? xr[k0 + c]     : 0.f;
        av[e].y = (k0 + c + 1 < K) ? xr[k0 + c + 1] : 0.f;
      } else {
        av[e] = *(const float2*)(xr + k0 + c);
      }
    }
    // ---- stage B tile (32 x 128): k-pairs per n -> packed ds_store_b32 --
    float bv0[8], bv1[8];
#pragma unroll
    for (int e = 0; e < 8; ++e) {
      int q = e * 256 + tid;                    // pair index 0..2047
      int n = q & 127;                          // 128 n per k-pair row
      int k = (q >> 7) * 2;                     // k-pair 0..15 -> k 0..30
      const float* wp = W + (size_t)(k0 + k) * N + n0 + n;
      if (KTAIL) {
        bv0[e] = (k0 + k     < K) ? wp[0] : 0.f;
        bv1[e] = (k0 + k + 1 < K) ? wp[N] : 0.f;
      } else {
        bv0[e] = wp[0];
        bv1[e] = wp[N];
      }
    }
#pragma unroll
    for (int e = 0; e < 8; ++e) {
      int q = e * 256 + tid;
      int r = q >> 4, c = (q & 15) * 2;
      v2h p; p[0] = (_Float16)av[e].x; p[1] = (_Float16)av[e].y;
      *(v2h*)&lsA[r * 32 + c] = p;              // ds_store_b32
    }
#pragma unroll
    for (int e = 0; e < 8; ++e) {
      int q = e * 256 + tid;
      int n = q & 127;
      int k = (q >> 7) * 2;
      v2h p; p[0] = (_Float16)bv0[e]; p[1] = (_Float16)bv1[e];
      *(v2h*)&lsB[n * 32 + k] = p;              // ds_store_b32 (transposed)
    }
    __syncthreads();

    // ---- A fragment: M row = wv*16 + l15; K = hi*8 + {0..7, 16..23} -----
    V16U af;
    {
      const _Float16* pa = &lsA[(wv * 16 + l15) * 32 + hi * 8];
      af.h[0] = *(const v8h*)(pa);
      af.h[1] = *(const v8h*)(pa + 16);
    }
#pragma unroll
    for (int nt = 0; nt < 8; ++nt) {
      // B fragment: N col = nt*16 + l15; K = hi*8 + {0..7, 16..23}
      const _Float16* pb = &lsB[(nt * 16 + l15) * 32 + hi * 8];
      V16U bf;
      bf.h[0] = *(const v8h*)(pb);
      bf.h[1] = *(const v8h*)(pb + 16);
      acc[nt] = __builtin_amdgcn_wmma_f32_16x16x32_f16(
          false, af.v, false, bf.v, (short)0, acc[nt], false, false);
    }
    __syncthreads();
  }

  // ---- epilogue: bias + activation + residual combine (straight-line) ---
#pragma unroll
  for (int nt = 0; nt < 8; ++nt) {
    int gn = n0 + nt * 16 + l15;
    float bb = bias[gn];
    F8U a; a.v = acc[nt];
    size_t base = (size_t)(m0 + wv * 16 + hi * 8) * N + gn;
    float r1v[8], r2v[8];
    if (COMB >= 1) {
#pragma unroll
      for (int r = 0; r < 8; ++r) r1v[r] = R1[base + (size_t)r * N];
    }
    if (COMB >= 2) {
#pragma unroll
      for (int r = 0; r < 8; ++r) r2v[r] = R2[base + (size_t)r * N];
    }
#pragma unroll
    for (int r = 0; r < 8; ++r) {
      float v = a.f[r] + bb;
      if      (ACT == 1) v = fmaxf(v, 0.f);
      else if (ACT == 2) v = 1.f / (1.f + __expf(-v));
      else if (ACT == 3) v = __expf(fminf(fmaxf(v, -50.f), 50.f));
      if      (COMB == 1) v += r1v[r];
      else if (COMB == 2) v += r1v[r] + r2v[r];
      else if (COMB == 3) v  = r1v[r] + r2v[r] * v;
      Y[base + (size_t)r * N] = v;
    }
  }
}

// one wave per row of 384; optional fused relu
__global__ __launch_bounds__(256)
void layernorm_k(const float* __restrict__ X, const float* __restrict__ g,
                 const float* __restrict__ b, float* __restrict__ Y,
                 int M, int relu_out) {
  int lane = threadIdx.x & 31;
  int row  = blockIdx.x * 8 + (threadIdx.x >> 5);
  if (row >= M) return;
  const float* xr = X + (size_t)row * DIM;
  float xv[12], s = 0.f, s2 = 0.f;
#pragma unroll
  for (int j = 0; j < 12; ++j) {
    xv[j] = xr[lane + j * 32];
    s += xv[j]; s2 += xv[j] * xv[j];
  }
#pragma unroll
  for (int o = 16; o > 0; o >>= 1) {
    s  += __shfl_xor(s,  o, 32);
    s2 += __shfl_xor(s2, o, 32);
  }
  float mean = s  * (1.f / DIM);
  float var  = s2 * (1.f / DIM) - mean * mean;
  float rs   = __frsqrt_rn(var + 1e-3f);
  float* yr  = Y + (size_t)row * DIM;
#pragma unroll
  for (int j = 0; j < 12; ++j) {
    int c = lane + j * 32;
    float v = (xv[j] - mean) * rs * g[c] + b[c];
    if (relu_out) v = fmaxf(v, 0.f);
    yr[c] = v;
  }
}

// replicate jnp argmax/argmin-first semantics of neighbors(kk, jj)
__global__ __launch_bounds__(256)
void neighbors_k(const int* __restrict__ kk, const int* __restrict__ jj,
                 int* __restrict__ ix, int* __restrict__ jx, int N) {
  __shared__ int skk[NEDGE];
  __shared__ int sjj[NEDGE];
  for (int i = threadIdx.x; i < N; i += 256) { skk[i] = kk[i]; sjj[i] = jj[i]; }
  __syncthreads();
  int n = blockIdx.x * 256 + threadIdx.x;
  if (n >= N) return;
  int myk = skk[n], myj = sjj[n];
  int bestv = -1,    bi = 0;       // argmax-first over prev_jj (invalid -> 0)
  int curv  = N + 1, ji = 0;       // argmin-first over next_jj (invalid -> N)
  for (int m = 0; m < N; ++m) {
    int same = (skk[m] == myk);
    int jm = sjj[m];
    int pv = (same && jm < myj) ? jm : 0;
    if (pv > bestv) { bestv = pv; bi = m; }
    int nv = (same && jm > myj) ? jm : N;
    if (nv < curv) { curv = nv; ji = m; }
  }
  ix[n] = bi;
  jx[n] = ji;
}

// deterministic segmented sums: one block (384 threads) per group g
// sums[0:G*384)      = max(sum_i w, 1e-9)
// sums[G*384:2G*384) = sum_i fx*w
__global__ __launch_bounds__(384)
void segsum_k(const int* __restrict__ idx, const float* __restrict__ FX,
              const float* __restrict__ Wt, float* __restrict__ sums, int N) {
  int g = blockIdx.x;
  int c = threadIdx.x;
  float sw = 0.f, sf = 0.f;
  for (int i = 0; i < N; ++i) {
    if (idx[i] == g) {
      size_t off = (size_t)i * DIM + c;
      float wv = Wt[off];
      sw += wv;
      sf += FX[off] * wv;
    }
  }
  sums[(size_t)g * DIM + c] = fmaxf(sw, 1e-9f);
  sums[(size_t)(gridDim.x + g) * DIM + c] = sf;
}

__global__ __launch_bounds__(256)
void aggy_k(const int* __restrict__ idx, const float* __restrict__ sums,
            float* __restrict__ Y, int N, int G) {
  int i = blockIdx.x * 256 + threadIdx.x;
  if (i >= N * DIM) return;
  int row = i / DIM, c = i - row * DIM;
  int g = idx[row];
  float den = sums[(size_t)g * DIM + c];
  float sf  = sums[(size_t)(G + g) * DIM + c];
  Y[i] = sf / den;
}

// output heads: d = relu(net)@Wd+bd ; w = sigmoid(relu(net)@Ww+bw); +ii*1e-10
__global__ __launch_bounds__(128)
void head_k(const float* __restrict__ net, const int* __restrict__ ii,
            const float* __restrict__ Wd, const float* __restrict__ bd,
            const float* __restrict__ Ww, const float* __restrict__ bw,
            float* __restrict__ dout, float* __restrict__ wout, int N) {
  int n = blockIdx.x * 128 + threadIdx.x;
  if (n >= N) return;
  const float* r = net + (size_t)n * DIM;
  float d0 = bd[0], d1 = bd[1], w0 = bw[0], w1 = bw[1];
  for (int k = 0; k < DIM; ++k) {
    float v = fmaxf(r[k], 0.f);
    d0 = fmaf(v, Wd[k * 2 + 0], d0);
    d1 = fmaf(v, Wd[k * 2 + 1], d1);
    w0 = fmaf(v, Ww[k * 2 + 0], w0);
    w1 = fmaf(v, Ww[k * 2 + 1], w1);
  }
  float bias = (float)ii[n] * 1e-10f;
  dout[n * 2 + 0] = d0 + bias;
  dout[n * 2 + 1] = d1 + bias;
  wout[n * 2 + 0] = 1.f / (1.f + __expf(-w0)) + bias;
  wout[n * 2 + 1] = 1.f / (1.f + __expf(-w1)) + bias;
}

extern "C" void kernel_launch(void* const* d_in, const int* in_sizes, int n_in,
                              void* d_out, int out_size, void* d_ws, size_t ws_size,
                              hipStream_t stream) {
  (void)in_sizes; (void)n_in; (void)out_size; (void)ws_size;
  const int N = NEDGE;
  const int KC = 2 * 49 * 3 * 3;   // 882

  // ---- inputs (setup_inputs insertion order) ----
  const float* net_in = (const float*)d_in[0];
  const float* inp    = (const float*)d_in[1];
  const float* corr   = (const float*)d_in[2];
  /* flow (d_in[3]) unused by reference */
  const int* ii = (const int*)d_in[4];
  const int* jj = (const int*)d_in[5];
  const int* kk = (const int*)d_in[6];
  // params: 25 (w,b) / (g,b) pairs in insertion order starting at index 7
  const float* P[50];
  for (int i = 0; i < 50; ++i) P[i] = (const float*)d_in[7 + i];
  #define PW(i) P[2*(i)]
  #define PB(i) P[2*(i)+1]
  // pair ids: 0 corr0, 1 corr1, 2 corr_ln, 3 corr2, 4 norm, 5 c1a, 6 c1b,
  // 7 c2a, 8 c2b, 9 agg_kk_f, 10 agg_kk_g, 11 agg_kk_h, 12 agg_ij_f,
  // 13 agg_ij_g, 14 agg_ij_h, 15 gru_ln1, 16 gr1_gate, 17 gr1_r1, 18 gr1_r2,
  // 19 gru_ln2, 20 gr2_gate, 21 gr2_r1, 22 gr2_r2, 23 d, 24 w

  // ---- workspace layout ----
  const size_t TSZ = (size_t)N * DIM;            // 1.57M floats per tensor
  float* W0   = (float*)d_ws;
  float* W1   = W0 + TSZ;
  float* W2   = W1 + TSZ;
  float* GT   = W2 + TSZ;                        // gate / exp-weights
  float* YB   = GT + TSZ;                        // aggregated y
  float* SUMS = YB + TSZ;                        // 2*512*384 floats max
  int*   ix   = (int*)(SUMS + 2 * 512 * DIM);
  int*   jx   = ix + N;

  float* netOut = (float*)d_out;                 // final net -> d_out[0:TSZ)
  float* dOut   = netOut + TSZ;                  // (N,2)
  float* wOut   = dOut + (size_t)N * 2;          // (N,2)

  dim3 blk(256);
  const dim3 gDD(DIM / 128, N / 128);            // 3 x 32 for DIM x DIM gemms

  #define GEMM(Gt, Kt, Ac, Cb, X, rm, Wm, bs, R1p, R2p, Yp, Kdim)            \
    gemm_f16_wmma<Gt, Kt, Ac, Cb><<<gDD, blk, 0, stream>>>(                  \
        X, rm, Wm, bs, R1p, R2p, Yp, N, Kdim, DIM)

  auto LN = [&](const float* X, int pid, float* Yp, int relu) {
    layernorm_k<<<dim3(N / 8), blk, 0, stream>>>(X, PW(pid), PB(pid), Yp, N, relu);
  };

  // ---- corr encoder: c = corr2( relu(ln( corr1( relu(corr0(corr)) ))) ) ----
  GEMM(0, 1, 1, 0, corr, nullptr, PW(0), PB(0), nullptr, nullptr, W0, KC);
  GEMM(0, 0, 0, 0, W0,   nullptr, PW(1), PB(1), nullptr, nullptr, W1, DIM);
  LN(W1, 2, W0, /*relu=*/1);
  // net = net_in + inp + corr2(c);  then layer_norm
  GEMM(0, 0, 0, 2, W0, nullptr, PW(3), PB(3), net_in, inp, W1, DIM);
  LN(W1, 4, W2, 0);                               // W2 = net_v1

  // ---- neighbor gathers ----
  neighbors_k<<<dim3(N / 256), blk, 0, stream>>>(kk, jj, ix, jx, N);
  GEMM(1, 0, 1, 0, W2, ix, PW(5), PB(5), nullptr, nullptr, W0, DIM);
  GEMM(0, 0, 0, 1, W0, nullptr, PW(6), PB(6), W2, nullptr, W1, DIM); // net_v2
  GEMM(1, 0, 1, 0, W1, jx, PW(7), PB(7), nullptr, nullptr, W0, DIM);
  GEMM(0, 0, 0, 1, W0, nullptr, PW(8), PB(8), W1, nullptr, W2, DIM); // net_v3

  // ---- soft_agg over kk (512 groups) ----
  GEMM(0, 0, 0, 0, W2, nullptr, PW(9),  PB(9),  nullptr, nullptr, W0, DIM); // fx
  GEMM(0, 0, 3, 0, W2, nullptr, PW(10), PB(10), nullptr, nullptr, GT, DIM); // w
  segsum_k<<<dim3(512), dim3(DIM), 0, stream>>>(kk, W0, GT, SUMS, N);
  aggy_k<<<dim3((N * DIM + 255) / 256), blk, 0, stream>>>(kk, SUMS, YB, N, 512);
  GEMM(0, 0, 0, 1, YB, nullptr, PW(11), PB(11), W2, nullptr, W1, DIM); // net_v4

  // ---- soft_agg over ii (64 groups) ----
  GEMM(0, 0, 0, 0, W1, nullptr, PW(12), PB(12), nullptr, nullptr, W0, DIM);
  GEMM(0, 0, 3, 0, W1, nullptr, PW(13), PB(13), nullptr, nullptr, GT, DIM);
  segsum_k<<<dim3(64), dim3(DIM), 0, stream>>>(ii, W0, GT, SUMS, N);
  aggy_k<<<dim3((N * DIM + 255) / 256), blk, 0, stream>>>(ii, SUMS, YB, N, 64);
  GEMM(0, 0, 0, 1, YB, nullptr, PW(14), PB(14), W1, nullptr, W2, DIM); // net_v5

  // ---- gated residual block 1 ----
  LN(W2, 15, W0, 0);                                                   // x
  GEMM(0, 0, 2, 0, W0, nullptr, PW(16), PB(16), nullptr, nullptr, GT, DIM);
  GEMM(0, 0, 1, 0, W0, nullptr, PW(17), PB(17), nullptr, nullptr, W1, DIM);
  GEMM(0, 0, 0, 3, W1, nullptr, PW(18), PB(18), W0, GT, W2, DIM);     // x+g*res

  // ---- gated residual block 2 (writes final net directly to d_out) ----
  LN(W2, 19, W0, 0);
  GEMM(0, 0, 2, 0, W0, nullptr, PW(20), PB(20), nullptr, nullptr, GT, DIM);
  GEMM(0, 0, 1, 0, W0, nullptr, PW(21), PB(21), nullptr, nullptr, W1, DIM);
  GEMM(0, 0, 0, 3, W1, nullptr, PW(22), PB(22), W0, GT, netOut, DIM); // net

  // ---- output heads ----
  head_k<<<dim3(N / 128), dim3(128), 0, stream>>>(netOut, ii, PW(23), PB(23),
                                                  PW(24), PB(24), dOut, wOut, N);
  #undef GEMM
  #undef PW
  #undef PB
}
